// DCABlock_57741540328117
// MI455X (gfx1250) — matrix operand order repaired and checked by reading, not compiled, not run on hardware
//
#include <hip/hip_runtime.h>

// ---------------------------------------------------------------------------
// DCA block for MI455X (gfx1250, wave32, WMMA).
// stats -> finalize -> WMMA pre-GEMM (async LDS staged) ->
// depthwise dilated conv+GLU -> SE linear -> WMMA post-GEMM (+bias+residual).
// ---------------------------------------------------------------------------

#define CCH   64            // input channels
#define H2    256           // pre expansion (2*HID)
#define HID   128
#define SPAT  262144        // 64*64*64
#define BATCH 2
#define EPS   1e-05f

#if defined(__has_builtin)
#if __has_builtin(__builtin_amdgcn_global_load_async_to_lds_b128) && \
    __has_builtin(__builtin_amdgcn_s_wait_asynccnt)
#define USE_ASYNC_LDS 1
#endif
#endif

typedef __attribute__((ext_vector_type(16))) __bf16 v16bf;
typedef __attribute__((ext_vector_type(8)))  float  v8f;
typedef __attribute__((ext_vector_type(4)))  float  v4f;
typedef __attribute__((ext_vector_type(4)))  int    i32x4;

union Frag {                 // 8 VGPRs worth of packed bf16 pairs
    unsigned int u[8];
    v16bf        v;
};

// ---- bf16 helpers ---------------------------------------------------------
__device__ __forceinline__ unsigned short f2bf(float f) {
    return (unsigned short)((__float_as_uint(f) + 0x8000u) >> 16);
}
__device__ __forceinline__ float bf2f(unsigned short h) {
    return __uint_as_float(((unsigned int)h) << 16);
}
// one v_perm_b32 packs both high halves: bytes {ul[2],ul[3],uh[2],uh[3]}
__device__ __forceinline__ unsigned int pack2bf(float lo, float hi) {
    unsigned int ul = __float_as_uint(lo) + 0x8000u;
    unsigned int uh = __float_as_uint(hi) + 0x8000u;
    return __builtin_amdgcn_perm(uh, ul, 0x07060302u);
}

// ---- WMMA fragment K-index maps (CDNA5 ISA 7.12.2, wave32) ----------------
// A 16x32 bf16: rows M=lane%16; VGPR v K pair: v<4 -> 2v else 16+2(v-4); +8*half
__device__ __forceinline__ int kpairA(int v, int half) {
    return (v < 4 ? 2 * v : 16 + 2 * (v - 4)) + 8 * half;
}
// B 32x16 bf16: cols N=lane%16; VGPR v K pair: v<4 -> 2v else 16+2(v-4); +8*half
__device__ __forceinline__ int kpairB(int v, int half) {
    return (v < 4 ? 2 * v : 16 + 2 * (v - 4)) + 8 * half;
}

// ===========================================================================
// K1: per-batch sum / sum-of-squares partial reduction (atomic accumulate)
// ===========================================================================
__global__ __launch_bounds__(256) void dca_stats_kernel(
    const float* __restrict__ x, float* __restrict__ stats)
{
    const int b = blockIdx.y;
    const size_t NB = (size_t)CCH * SPAT;
    const float* xb = x + (size_t)b * NB;

    float s = 0.f, ss = 0.f;
    for (size_t i = (size_t)blockIdx.x * blockDim.x + threadIdx.x; i < NB;
         i += (size_t)gridDim.x * blockDim.x) {
        float v = xb[i];
        s += v;
        ss += v * v;
    }
    __shared__ float sh1[256], sh2[256];
    sh1[threadIdx.x] = s;
    sh2[threadIdx.x] = ss;
    __syncthreads();
    for (int off = 128; off > 0; off >>= 1) {
        if ((int)threadIdx.x < off) {
            sh1[threadIdx.x] += sh1[threadIdx.x + off];
            sh2[threadIdx.x] += sh2[threadIdx.x + off];
        }
        __syncthreads();
    }
    if (threadIdx.x == 0) {
        atomicAdd(&stats[b * 2 + 0], sh1[0]);
        atomicAdd(&stats[b * 2 + 1], sh2[0]);
    }
}

// ===========================================================================
// K2: finalize mean / inverse sigma per batch
// ===========================================================================
__global__ void dca_finalize_stats_kernel(const float* __restrict__ stats,
                                          float* __restrict__ musig)
{
    if (threadIdx.x == 0 && blockIdx.x == 0) {
        const float invN = 1.0f / ((float)CCH * (float)SPAT);
        for (int b = 0; b < BATCH; ++b) {
            float mu  = stats[b * 2 + 0] * invN;
            float var = stats[b * 2 + 1] * invN - mu * mu;
            musig[b * 2 + 0] = mu;
            musig[b * 2 + 1] = rsqrtf(var + EPS);
        }
    }
}

// ===========================================================================
// K3: normalize + pre 1x1 conv as WMMA GEMM.  y[o,s] = pre_w[o,:] . xn[:,s]
//     Block: 256 threads = 8 waves; N-tile = 128 voxels (one 16-voxel
//     subtile per wave, so x is loaded exactly once); each wave loops all
//     16 M-tiles.  K=64 -> 2 chained bf16 WMMAs per tile.
//     x tile staged into LDS (async-to-LDS when available); pre_w staged
//     into LDS once per block as packed bf16 pairs (bank-padded).
// ===========================================================================
__global__ __launch_bounds__(256) void dca_pre_gemm_kernel(
    const float* __restrict__ x, const float* __restrict__ gamma,
    const float* __restrict__ beta, const float* __restrict__ pre_w,
    const float* __restrict__ pre_b, const float* __restrict__ musig,
    unsigned short* __restrict__ yb)
{
    const int b    = blockIdx.y;
    const int s0   = blockIdx.x * 128;
    const int lane = threadIdx.x & 31;
    const int wave = threadIdx.x >> 5;       // 0..7 -> N subtile
    const int n    = lane & 15;
    const int half = lane >> 4;
    const int s    = s0 + wave * 16 + n;     // this wave's voxel column

    __shared__ float        xs[CCH][128];    // fp32 x tile (32 KB)
    __shared__ unsigned int aw[H2][34];      // packed bf16 pre_w, padded (34 KB)

    const float* xb = x + (size_t)b * CCH * SPAT;

    // ---- stage x tile: 64 ch x 128 voxels, 16B vectors -------------------
#ifdef USE_ASYNC_LDS
    for (int e = threadIdx.x; e < CCH * 32; e += 256) {   // 2048 float4s
        const int k  = e >> 5;
        const int j4 = (e & 31) * 4;
        const float* gp = xb + (size_t)k * SPAT + s0 + j4;
        __builtin_amdgcn_global_load_async_to_lds_b128(
            (__attribute__((address_space(1))) i32x4*)gp,
            (__attribute__((address_space(3))) i32x4*)&xs[k][j4], 0, 0);
    }
#else
    for (int e = threadIdx.x; e < CCH * 32; e += 256) {
        const int k  = e >> 5;
        const int j4 = (e & 31) * 4;
        *(v4f*)&xs[k][j4] = *(const v4f*)(xb + (size_t)k * SPAT + s0 + j4);
    }
#endif

    // ---- stage pre_w as packed bf16 pairs: aw[m][k/2] --------------------
    for (int e = threadIdx.x; e < H2 * 32; e += 256) {
        const int m  = e >> 5;
        const int kk = e & 31;
        aw[m][kk] = pack2bf(pre_w[m * CCH + 2 * kk], pre_w[m * CCH + 2 * kk + 1]);
    }

#ifdef USE_ASYNC_LDS
    __builtin_amdgcn_s_wait_asynccnt(0);
#endif
    __syncthreads();

    const float mu = musig[b * 2 + 0];
    const float rs = musig[b * 2 + 1];
    const int   j  = wave * 16 + n;

    // ---- B fragments: normalized activations for this wave's 16 voxels --
    Frag Bf[2];
    for (int kf = 0; kf < 2; ++kf) {
        #pragma unroll
        for (int v = 0; v < 8; ++v) {
            const int k  = kpairB(v, half) + 32 * kf;
            const float sc0 = rs * gamma[k],     sh0 = beta[k]     - mu * sc0;
            const float sc1 = rs * gamma[k + 1], sh1 = beta[k + 1] - mu * sc1;
            Bf[kf].u[v] = pack2bf(xs[k][j] * sc0 + sh0, xs[k + 1][j] * sc1 + sh1);
        }
    }

    // ---- loop all 16 M-tiles, A from LDS, 2 chained WMMAs each -----------
    for (int mt = 0; mt < 16; ++mt) {
        const int m = mt * 16 + n;
        Frag Af[2];
        for (int kf = 0; kf < 2; ++kf) {
            #pragma unroll
            for (int v = 0; v < 8; ++v)
                Af[kf].u[v] = aw[m][(kpairA(v, half) + 32 * kf) >> 1];
        }
        v8f acc;
        #pragma unroll
        for (int v = 0; v < 8; ++v)
            acc[v] = pre_b[mt * 16 + v + 8 * half];        // bias in C
        acc = __builtin_amdgcn_wmma_f32_16x16x32_bf16(
            false, Af[0].v, false, Bf[0].v, (short)0, acc, false, false);
        acc = __builtin_amdgcn_wmma_f32_16x16x32_bf16(
            false, Af[1].v, false, Bf[1].v, (short)0, acc, false, false);
        #pragma unroll
        for (int v = 0; v < 8; ++v) {
            const int o = mt * 16 + v + 8 * half;
            yb[((size_t)b * H2 + o) * SPAT + s] = f2bf(acc[v]);
        }
    }
}

// ===========================================================================
// K4: depthwise dilated 3x3x3 conv (dil=3, pad=3) + GLU gate + pooled sums.
// ===========================================================================
__global__ __launch_bounds__(256) void dca_dconv_glu_kernel(
    const unsigned short* __restrict__ yb, const float* __restrict__ ddc_w,
    const float* __restrict__ ddc_b, unsigned short* __restrict__ gb,
    float* __restrict__ pooled)
{
    const int b  = blockIdx.z;
    const int ha = blockIdx.y;                    // gate channel 0..127
    const int s  = blockIdx.x * 256 + threadIdx.x;
    const int w  = s & 63;
    const int h  = (s >> 6) & 63;
    const int d  = s >> 12;

    float acc[2];
    #pragma unroll
    for (int p = 0; p < 2; ++p) {
        const int ch = ha + p * HID;
        const unsigned short* ybc = yb + ((size_t)b * H2 + ch) * SPAT;
        const float* wc = ddc_w + ch * 27;
        float a = ddc_b[ch];
        #pragma unroll
        for (int kd = 0; kd < 3; ++kd) {
            const int dd = d + (kd - 1) * 3;
            if ((unsigned)dd >= 64u) continue;
            #pragma unroll
            for (int kh = 0; kh < 3; ++kh) {
                const int hh = h + (kh - 1) * 3;
                if ((unsigned)hh >= 64u) continue;
                #pragma unroll
                for (int kw = 0; kw < 3; ++kw) {
                    const int ww = w + (kw - 1) * 3;
                    if ((unsigned)ww >= 64u) continue;
                    a += bf2f(ybc[(((dd << 6) + hh) << 6) + ww]) *
                         wc[kd * 9 + kh * 3 + kw];
                }
            }
        }
        acc[p] = a;
    }
    const float g = acc[0] * acc[1];
    gb[((size_t)b * HID + ha) * SPAT + s] = f2bf(g);

    __shared__ float sh[256];
    sh[threadIdx.x] = g;
    __syncthreads();
    for (int off = 128; off > 0; off >>= 1) {
        if ((int)threadIdx.x < off) sh[threadIdx.x] += sh[threadIdx.x + off];
        __syncthreads();
    }
    if (threadIdx.x == 0) atomicAdd(&pooled[b * HID + ha], sh[0]);
}

// ===========================================================================
// K5: SE linear  attn[b,o] = sca_b[o] + mean_pool[b,:] . sca_w[o,:]
// ===========================================================================
__global__ void dca_sca_kernel(const float* __restrict__ pooled,
                               const float* __restrict__ sca_w,
                               const float* __restrict__ sca_b,
                               float* __restrict__ attn)
{
    const int t = threadIdx.x;
    if (t >= BATCH * HID) return;
    const int b = t >> 7;
    const int o = t & 127;
    const float inv = 1.0f / (float)SPAT;
    float a = sca_b[o];
    for (int h = 0; h < HID; ++h)
        a += (pooled[b * HID + h] * inv) * sca_w[o * HID + h];
    attn[b * HID + o] = a;
}

// ===========================================================================
// K6: post 1x1 conv as WMMA GEMM: out[c,s] = post_w[c,:].(g[:,s]*attn[:])
//     + post_b[c] + residual.  Block: 256 threads = 8 waves; N-tile = 128
//     voxels (one subtile per wave -> g read once); each wave loops the 4
//     M-tiles.  K=128 -> 4 chained bf16 WMMAs.  post_w staged in LDS.
// ===========================================================================
__global__ __launch_bounds__(256) void dca_post_gemm_kernel(
    const unsigned short* __restrict__ gb, const float* __restrict__ post_w,
    const float* __restrict__ post_b, const float* __restrict__ attn,
    const float* __restrict__ x, float* __restrict__ out)
{
    const int b    = blockIdx.y;
    const int s0   = blockIdx.x * 128;
    const int lane = threadIdx.x & 31;
    const int wave = threadIdx.x >> 5;
    const int n    = lane & 15;
    const int half = lane >> 4;
    const int s    = s0 + wave * 16 + n;

    __shared__ unsigned int aw[CCH][65];     // packed bf16 post_w, padded

    for (int e = threadIdx.x; e < CCH * 64; e += 256) {
        const int m  = e >> 6;
        const int kk = e & 63;
        aw[m][kk] = pack2bf(post_w[m * HID + 2 * kk], post_w[m * HID + 2 * kk + 1]);
    }
    __syncthreads();

    const float* attb = attn + b * HID;
    const unsigned short* gbb = gb + (size_t)b * HID * SPAT;

    Frag Bf[4];
    for (int kf = 0; kf < 4; ++kf) {
        #pragma unroll
        for (int v = 0; v < 8; ++v) {
            const int hh = kpairB(v, half) + 32 * kf;
            float g0 = bf2f(gbb[(size_t)hh * SPAT + s]) * attb[hh];
            float g1 = bf2f(gbb[(size_t)(hh + 1) * SPAT + s]) * attb[hh + 1];
            Bf[kf].u[v] = pack2bf(g0, g1);
        }
    }

    for (int mt = 0; mt < 4; ++mt) {
        const int m = mt * 16 + n;
        Frag Af[4];
        for (int kf = 0; kf < 4; ++kf) {
            #pragma unroll
            for (int v = 0; v < 8; ++v)
                Af[kf].u[v] = aw[m][(kpairA(v, half) + 32 * kf) >> 1];
        }
        v8f acc;
        #pragma unroll
        for (int v = 0; v < 8; ++v)
            acc[v] = post_b[mt * 16 + v + 8 * half];
        #pragma unroll
        for (int kf = 0; kf < 4; ++kf)
            acc = __builtin_amdgcn_wmma_f32_16x16x32_bf16(
                false, Af[kf].v, false, Bf[kf].v, (short)0, acc, false, false);
        #pragma unroll
        for (int v = 0; v < 8; ++v) {
            const int c = mt * 16 + v + 8 * half;
            const size_t idx = ((size_t)b * CCH + c) * SPAT + s;
            out[idx] = acc[v] + x[idx];
        }
    }
}

// ===========================================================================
// Launcher
// ===========================================================================
extern "C" void kernel_launch(void* const* d_in, const int* in_sizes, int n_in,
                              void* d_out, int out_size, void* d_ws, size_t ws_size,
                              hipStream_t stream)
{
    (void)in_sizes; (void)n_in; (void)out_size; (void)ws_size;

    const float* x        = (const float*)d_in[0];
    const float* gn_gamma = (const float*)d_in[1];
    const float* gn_beta  = (const float*)d_in[2];
    const float* pre_w    = (const float*)d_in[3];
    const float* pre_b    = (const float*)d_in[4];
    const float* ddc_w    = (const float*)d_in[5];
    const float* ddc_b    = (const float*)d_in[6];
    const float* sca_w    = (const float*)d_in[7];
    const float* sca_b    = (const float*)d_in[8];
    const float* post_w   = (const float*)d_in[9];
    const float* post_b   = (const float*)d_in[10];
    float*       out      = (float*)d_out;

    // workspace layout
    char* ws = (char*)d_ws;
    float* stats  = (float*)(ws + 0);          // 2*2 f32
    float* musig  = (float*)(ws + 64);         // 2*2 f32
    float* pooled = (float*)(ws + 128);        // 2*128 f32
    float* attn   = (float*)(ws + 1280);       // 2*128 f32
    unsigned short* yb = (unsigned short*)(ws + 4096);                    // 256 MB bf16
    unsigned short* gb = (unsigned short*)(ws + 4096 +
                          (size_t)BATCH * H2 * SPAT * sizeof(unsigned short)); // 128 MB bf16

    // zero accumulators (graph-capture-safe memset node)
    (void)hipMemsetAsync(d_ws, 0, 2304, stream);

    // K1/K2: layer-norm statistics
    dca_stats_kernel<<<dim3(1024, BATCH), 256, 0, stream>>>(x, stats);
    dca_finalize_stats_kernel<<<1, 32, 0, stream>>>(stats, musig);

    // K3: normalize + pre GEMM (WMMA), y bf16
    dca_pre_gemm_kernel<<<dim3(SPAT / 128, BATCH), 256, 0, stream>>>(
        x, gn_gamma, gn_beta, pre_w, pre_b, musig, yb);

    // K4: depthwise dilated conv + GLU + pooled sums
    dca_dconv_glu_kernel<<<dim3(SPAT / 256, HID, BATCH), 256, 0, stream>>>(
        yb, ddc_w, ddc_b, gb, pooled);

    // K5: SE linear
    dca_sca_kernel<<<1, 256, 0, stream>>>(pooled, sca_w, sca_b, attn);

    // K6: post GEMM (WMMA) + bias + residual
    dca_post_gemm_kernel<<<dim3(SPAT / 128, BATCH), 256, 0, stream>>>(
        gb, post_w, post_b, attn, x, out);
}
